// PointPillars_83537113907653
// MI455X (gfx1250) — compile-verified, matrix operand order checked by reading, not run on hardware
//
#include <hip/hip_runtime.h>
#include <hip/hip_bf16.h>
#include <math.h>

// ---------------- model constants ----------------
#define BSZ      4
#define NPIL     48000
#define NPTS     32
#define XL       432
#define YL       496
#define CPE      64
#define NECK     128
#define HF       248     // final feature H (= YL/2)
#define WF       216     // final feature W (= XL/2)
#define EPSB     1e-3f

typedef __attribute__((ext_vector_type(16))) __bf16 v16bf;
typedef __attribute__((ext_vector_type(8)))  __bf16 v8bf;
typedef __attribute__((ext_vector_type(8)))  float  v8f;

// K index held by element e (0..15) of a lane in half `h` (lane>>4) for the
// 16-bit 16x32 A fragment (CDNA5 ISA 7.12.2). For e<8:  k = 8h + e.
// For e>=8: k = 16 + 8h + (e-8). B (32x16) assumed mirrored (lane = column).
__device__ __forceinline__ int kmap16(int e, int h) {
    const int vv = e >> 1, lo = e & 1;
    const int base = (vv < 4) ? (vv * 2) : (16 + (vv - 4) * 2);
    return base + h * 8 + lo;
}

#define CAT16(lo, hi) __builtin_shufflevector(lo, hi, 0,1,2,3,4,5,6,7,8,9,10,11,12,13,14,15)
#define WMMA_BF16(a, b, c) __builtin_amdgcn_wmma_f32_16x16x32_bf16( \
        false, a, false, b, (short)0, c, false, false)

// ---------------- pillar encoder (VALU; K=9 too small for WMMA) ----------------
__global__ void __launch_bounds__(64)
pe_encode(const float* __restrict__ pil, const int* __restrict__ coors,
          const int* __restrict__ npp, const float* __restrict__ pw,
          const float* __restrict__ g, const float* __restrict__ b,
          const float* __restrict__ m, const float* __restrict__ v,
          float* __restrict__ pooled)
{
    __shared__ float sp[NPTS * 4];
    const int p = blockIdx.x, t = threadIdx.x;
    sp[t]      = pil[(size_t)p * NPTS * 4 + t];
    sp[t + 64] = pil[(size_t)p * NPTS * 4 + 64 + t];
    __syncthreads();

    const int   np  = npp[p];
    const float inv = 1.f / (float)np;
    float cxs = 0.f, cys = 0.f, czs = 0.f;
    for (int n = 0; n < NPTS; ++n) { cxs += sp[n*4]; cys += sp[n*4+1]; czs += sp[n*4+2]; }
    cxs *= inv; cys *= inv; czs *= inv;

    const float cx = (float)coors[p*4+1] * 0.16f + 0.08f;
    const float cy = (float)coors[p*4+2] * 0.16f - 39.6f;

    float wv[9];
#pragma unroll
    for (int j = 0; j < 9; ++j) wv[j] = pw[t * 9 + j];
    const float s  = g[t] * rsqrtf(v[t] + EPSB);
    const float sh = b[t] - m[t] * s;

    float best = 0.f;
    for (int n = 0; n < NPTS; ++n) {
        const float px = sp[n*4], py = sp[n*4+1], pz = sp[n*4+2], pr = sp[n*4+3];
        const float xo = px - cx, yo = py - cy;
        float h = xo*wv[0] + yo*wv[1] + pz*wv[2] + pr*wv[3]
                + (px-cxs)*wv[4] + (py-cys)*wv[5] + (pz-czs)*wv[6]
                + xo*wv[7] + yo*wv[8];
        h = (n < np) ? h : 0.f;           // mask before BN; bn(0) still feeds the max
        h = fmaxf(h * s + sh, 0.f);
        best = fmaxf(best, h);
    }
    pooled[(size_t)p * CPE + t] = best;
}

// ---------------- deterministic scatter into bf16 NHWC canvas ----------------
__global__ void scatter_argmax(const int* __restrict__ coors, int* __restrict__ map)
{
    const int p = blockIdx.x * blockDim.x + threadIdx.x;
    if (p >= NPIL) return;
    const int b = coors[p*4], xi = coors[p*4+1], yi = coors[p*4+2];
    atomicMax(&map[(b * XL + xi) * YL + yi], p);
}

__global__ void scatter_write(const int* __restrict__ coors, const int* __restrict__ map,
                              const float* __restrict__ pooled, __bf16* __restrict__ canvas)
{
    const int idx = blockIdx.x * blockDim.x + threadIdx.x;
    const int p = idx >> 6, c = idx & 63;
    if (p >= NPIL) return;
    const int b = coors[p*4], xi = coors[p*4+1], yi = coors[p*4+2];
    if (map[(b * XL + xi) * YL + yi] != p) return;
    // canvas is NHWC: (B, YL, XL, CPE)
    canvas[(((size_t)b * YL + yi) * XL + xi) * CPE + c] = (__bf16)pooled[(size_t)p * CPE + c];
}

// ---------------- weight packing into WMMA B-fragment register image ----------
// dst[((kc*ntiles + nt)*32 + lane)*16 + e] = w element for (k = kc*32+kmap16, n = nt*16+lane%16)
__global__ void pack_conv_w(const float* __restrict__ w, __bf16* __restrict__ dst,
                            int cin, int cinShift, int cout, int ntiles, int total)
{
    const int idx = blockIdx.x * blockDim.x + threadIdx.x;
    if (idx >= total) return;
    const int e    = idx & 15;
    const int lane = (idx >> 4) & 31;
    const int rest = idx >> 9;
    const int nt   = rest % ntiles;
    const int kc   = rest / ntiles;
    const int k    = kc * 32 + kmap16(e, lane >> 4);
    const int n    = nt * 16 + (lane & 15);
    const int tap  = k >> cinShift;
    const int c    = k & (cin - 1);
    dst[idx] = (n < cout) ? (__bf16)w[((size_t)n * cin + c) * 9 + tap] : (__bf16)0.f;
}

// deconv weights (cin,128,up,up): dst[(((phase*KC)+kc)*8 + nt)*512 + lane*16 + e]
__global__ void pack_dec_w(const float* __restrict__ w, __bf16* __restrict__ dst,
                           int cin, int upup, int total)
{
    const int idx = blockIdx.x * blockDim.x + threadIdx.x;
    if (idx >= total) return;
    const int e    = idx & 15;
    const int lane = (idx >> 4) & 31;
    int rest = idx >> 9;
    const int nt = rest & 7;  rest >>= 3;
    const int kcn = cin >> 5;
    const int kc = rest % kcn;
    const int phase = rest / kcn;
    const int c = kc * 32 + kmap16(e, lane >> 4);
    const int n = nt * 16 + (lane & 15);
    dst[idx] = (__bf16)w[((size_t)c * NECK + n) * upup + phase];
}

// head weights (cout, K): dst[(kc*ntiles + nt)*512 + lane*16 + e], zero-padded in n
__global__ void pack_fc_w(const float* __restrict__ w, __bf16* __restrict__ dst,
                          int K, int cout, int ntiles, int total)
{
    const int idx = blockIdx.x * blockDim.x + threadIdx.x;
    if (idx >= total) return;
    const int e    = idx & 15;
    const int lane = (idx >> 4) & 31;
    const int rest = idx >> 9;
    const int nt   = rest % ntiles;
    const int kc   = rest / ntiles;
    const int c    = kc * 32 + kmap16(e, lane >> 4);
    const int n    = nt * 16 + (lane & 15);
    dst[idx] = (n < cout) ? (__bf16)w[(size_t)n * K + c] : (__bf16)0.f;
}

// ---------------- 3x3 conv + BN + ReLU, bf16 NHWC, 2x2-blocked WMMA ---------
template<int CIN, int COUT, int STRIDE>
__global__ void __launch_bounds__(256)
conv3x3_wmma(const __bf16* __restrict__ x, const __bf16* __restrict__ wp,
             const float* __restrict__ bng, const float* __restrict__ bnb,
             const float* __restrict__ bnm, const float* __restrict__ bnv,
             __bf16* __restrict__ y, int Hin, int Win, int Hout, int Wout)
{
    constexpr int KC = (CIN * 9) / 32;       // K chunks (exact)
    constexpr int NT = COUT / 16;
    constexpr int NG = NT / 2;               // N pair groups
    const int lane = threadIdx.x & 31;
    const int half = lane >> 4;
    const int l16  = lane & 15;
    const int gw   = blockIdx.x * (blockDim.x >> 5) + (threadIdx.x >> 5);

    const int Mtot   = BSZ * Hout * Wout;    // exact multiple of 16
    const int Mtiles = Mtot >> 4;
    const int mPairs = (Mtiles + 1) >> 1;
    if (gw >= mPairs * NG) return;
    const int mPair = gw % mPairs;
    const int nPair = gw / mPairs;
    const int mTile0 = mPair * 2;
    const int mTile1 = (mTile0 + 1 < Mtiles) ? (mTile0 + 1) : mTile0;  // clamp
    const int nTile0 = nPair * 2;

    int ow0, oh0, bi0, ow1, oh1, bi1;
    { int t = mTile0 * 16 + l16; ow0 = t % Wout; t /= Wout; oh0 = t % Hout; bi0 = t / Hout; }
    { int t = mTile1 * 16 + l16; ow1 = t % Wout; t /= Wout; oh1 = t % Hout; bi1 = t / Hout; }

    const __bf16* blane = wp + ((size_t)nTile0 * 32 + lane) * 16;

    v8f acc00 = {}, acc01 = {}, acc10 = {}, acc11 = {};
#pragma unroll 2
    for (int kc = 0; kc < KC; ++kc) {
        const int kt    = kc * 32;
        const int tap   = kt / CIN;          // whole chunk in one tap
        const int cbase = kt & (CIN - 1);
        const int ky = tap / 3, kx = tap - ky * 3;

        v8bf a0l = {}, a0h = {}, a1l = {}, a1h = {};
        {
            const int ih = oh0 * STRIDE - 1 + ky, iw = ow0 * STRIDE - 1 + kx;
            if ((unsigned)ih < (unsigned)Hin && (unsigned)iw < (unsigned)Win) {
                const __bf16* ap = x + ((size_t)(bi0 * Hin + ih) * Win + iw) * CIN
                                 + cbase + half * 8;
                a0l = *(const v8bf*)(ap);
                a0h = *(const v8bf*)(ap + 16);
            }
        }
        {
            const int ih = oh1 * STRIDE - 1 + ky, iw = ow1 * STRIDE - 1 + kx;
            if ((unsigned)ih < (unsigned)Hin && (unsigned)iw < (unsigned)Win) {
                const __bf16* ap = x + ((size_t)(bi1 * Hin + ih) * Win + iw) * CIN
                                 + cbase + half * 8;
                a1l = *(const v8bf*)(ap);
                a1h = *(const v8bf*)(ap + 16);
            }
        }
        const __bf16* bp = blane + (size_t)kc * (NT * 512);
        const v8bf b0l = *(const v8bf*)(bp);
        const v8bf b0h = *(const v8bf*)(bp + 8);
        const v8bf b1l = *(const v8bf*)(bp + 512);
        const v8bf b1h = *(const v8bf*)(bp + 520);
        __builtin_prefetch((const void*)(bp + (size_t)NT * 512), 0, 1);

        const v16bf a0 = CAT16(a0l, a0h);
        const v16bf a1 = CAT16(a1l, a1h);
        const v16bf b0 = CAT16(b0l, b0h);
        const v16bf b1 = CAT16(b1l, b1h);
        acc00 = WMMA_BF16(a0, b0, acc00);
        acc01 = WMMA_BF16(a0, b1, acc01);
        acc10 = WMMA_BF16(a1, b0, acc10);
        acc11 = WMMA_BF16(a1, b1, acc11);
    }

    const int n0 = nTile0 * 16 + l16, n1 = n0 + 16;
    const float s0  = bng[n0] * rsqrtf(bnv[n0] + EPSB);
    const float sh0 = bnb[n0] - bnm[n0] * s0;
    const float s1  = bng[n1] * rsqrtf(bnv[n1] + EPSB);
    const float sh1 = bnb[n1] - bnm[n1] * s1;
    const bool dup = (mTile1 == mTile0);
#pragma unroll
    for (int r = 0; r < 8; ++r) {
        const int mg0 = mTile0 * 16 + half * 8 + r;   // D: M = r + 8*half, N = l16
        y[(size_t)mg0 * COUT + n0] = (__bf16)fmaxf(acc00[r] * s0 + sh0, 0.f);
        y[(size_t)mg0 * COUT + n1] = (__bf16)fmaxf(acc01[r] * s1 + sh1, 0.f);
        if (!dup) {
            const int mg1 = mTile1 * 16 + half * 8 + r;
            y[(size_t)mg1 * COUT + n0] = (__bf16)fmaxf(acc10[r] * s0 + sh0, 0.f);
            y[(size_t)mg1 * COUT + n1] = (__bf16)fmaxf(acc11[r] * s1 + sh1, 0.f);
        }
    }
}

// ---------------- transposed conv (k==s): per-phase 1x1 GEMM, 2x2-blocked ----
template<int CIN, int UP>
__global__ void __launch_bounds__(256)
deconv_wmma(const __bf16* __restrict__ x, const __bf16* __restrict__ wp,
            const float* __restrict__ bng, const float* __restrict__ bnb,
            const float* __restrict__ bnm, const float* __restrict__ bnv,
            __bf16* __restrict__ feat, int Hin, int Win, int chOff)
{
    constexpr int KC = CIN / 32;
    constexpr int NT = NECK / 16;            // 8
    constexpr int NG = NT / 2;               // 4
    const int lane = threadIdx.x & 31;
    const int half = lane >> 4;
    const int l16  = lane & 15;
    const int gw   = blockIdx.x * (blockDim.x >> 5) + (threadIdx.x >> 5);

    const int Mtot   = BSZ * Hin * Win;      // exact multiple of 16
    const int Mtiles = Mtot >> 4;
    const int mPairs = (Mtiles + 1) >> 1;
    if (gw >= mPairs * NG) return;
    const int mPair = gw % mPairs;
    const int nPair = gw / mPairs;
    const int phase = blockIdx.y;
    const int mTile0 = mPair * 2;
    const int mTile1 = (mTile0 + 1 < Mtiles) ? (mTile0 + 1) : mTile0;
    const int nTile0 = nPair * 2;

    const __bf16* ap0 = x + (size_t)(mTile0 * 16 + l16) * CIN + half * 8;
    const __bf16* ap1 = x + (size_t)(mTile1 * 16 + l16) * CIN + half * 8;
    const __bf16* bp0 = wp + ((((size_t)phase * KC) * NT + nTile0) * 32 + lane) * 16;

    v8f acc00 = {}, acc01 = {}, acc10 = {}, acc11 = {};
#pragma unroll 2
    for (int kc = 0; kc < KC; ++kc) {
        const v8bf a0l = *(const v8bf*)(ap0 + kc * 32);
        const v8bf a0h = *(const v8bf*)(ap0 + kc * 32 + 16);
        const v8bf a1l = *(const v8bf*)(ap1 + kc * 32);
        const v8bf a1h = *(const v8bf*)(ap1 + kc * 32 + 16);
        const __bf16* bp = bp0 + (size_t)kc * (NT * 512);
        const v8bf b0l = *(const v8bf*)(bp);
        const v8bf b0h = *(const v8bf*)(bp + 8);
        const v8bf b1l = *(const v8bf*)(bp + 512);
        const v8bf b1h = *(const v8bf*)(bp + 520);
        const v16bf a0 = CAT16(a0l, a0h);
        const v16bf a1 = CAT16(a1l, a1h);
        const v16bf b0 = CAT16(b0l, b0h);
        const v16bf b1 = CAT16(b1l, b1h);
        acc00 = WMMA_BF16(a0, b0, acc00);
        acc01 = WMMA_BF16(a0, b1, acc01);
        acc10 = WMMA_BF16(a1, b0, acc10);
        acc11 = WMMA_BF16(a1, b1, acc11);
    }

    const int dy = phase / UP, dx = phase - dy * UP;
    const int n0 = nTile0 * 16 + l16, n1 = n0 + 16;
    const float s0  = bng[n0] * rsqrtf(bnv[n0] + EPSB);
    const float sh0 = bnb[n0] - bnm[n0] * s0;
    const float s1  = bng[n1] * rsqrtf(bnv[n1] + EPSB);
    const float sh1 = bnb[n1] - bnm[n1] * s1;
    const bool dup = (mTile1 == mTile0);
#pragma unroll
    for (int r = 0; r < 8; ++r) {
        {
            int t = mTile0 * 16 + half * 8 + r;
            const int x2 = t % Win; t /= Win;
            const int y2 = t % Hin;
            const int b2 = t / Hin;
            const size_t o = (((size_t)b2 * HF + y2 * UP + dy) * WF + x2 * UP + dx)
                           * (3 * NECK) + chOff;
            feat[o + n0] = (__bf16)fmaxf(acc00[r] * s0 + sh0, 0.f);
            feat[o + n1] = (__bf16)fmaxf(acc01[r] * s1 + sh1, 0.f);
        }
        if (!dup) {
            int t = mTile1 * 16 + half * 8 + r;
            const int x2 = t % Win; t /= Win;
            const int y2 = t % Hin;
            const int b2 = t / Hin;
            const size_t o = (((size_t)b2 * HF + y2 * UP + dy) * WF + x2 * UP + dx)
                           * (3 * NECK) + chOff;
            feat[o + n0] = (__bf16)fmaxf(acc10[r] * s0 + sh0, 0.f);
            feat[o + n1] = (__bf16)fmaxf(acc11[r] * s1 + sh1, 0.f);
        }
    }
}

// ---------------- head 1x1 conv + bias: bf16 NHWC in, f32 NCHW out -----------
template<int COUT>
__global__ void __launch_bounds__(256)
head_wmma(const __bf16* __restrict__ feat, const __bf16* __restrict__ wp,
          const float* __restrict__ bias, float* __restrict__ out)
{
    constexpr int K  = 3 * NECK;             // 384
    constexpr int KC = K / 32;               // 12
    constexpr int NT = (COUT + 15) / 16;
    const int lane = threadIdx.x & 31;
    const int half = lane >> 4;
    const int l16  = lane & 15;
    const int gw   = blockIdx.x * (blockDim.x >> 5) + (threadIdx.x >> 5);

    const int Mtot   = BSZ * HF * WF;        // exact multiple of 16
    const int Mtiles = Mtot >> 4;
    if (gw >= Mtiles * NT) return;
    const int mTile = gw % Mtiles;
    const int nTile = gw / Mtiles;

    const __bf16* ap0 = feat + (size_t)(mTile * 16 + l16) * K + half * 8;
    const __bf16* bp0 = wp + ((size_t)nTile * 32 + lane) * 16;

    v8f acc = {};
#pragma unroll 4
    for (int kc = 0; kc < KC; ++kc) {
        const v8bf alo = *(const v8bf*)(ap0 + kc * 32);
        const v8bf ahi = *(const v8bf*)(ap0 + kc * 32 + 16);
        const __bf16* bp = bp0 + (size_t)kc * (NT * 512);
        const v8bf blo = *(const v8bf*)(bp);
        const v8bf bhi = *(const v8bf*)(bp + 8);
        const v16bf av = CAT16(alo, ahi);
        const v16bf bv = CAT16(blo, bhi);
        acc = WMMA_BF16(av, bv, acc);
    }

    const int n = nTile * 16 + l16;
    if (n >= COUT) return;
    const float bb = bias[n];
#pragma unroll
    for (int r = 0; r < 8; ++r) {
        int t = mTile * 16 + half * 8 + r;
        const int x2 = t % WF; t /= WF;
        const int y2 = t % HF;
        const int b2 = t / HF;
        out[(((size_t)b2 * COUT + n) * HF + y2) * WF + x2] = acc[r] + bb;
    }
}

// ---------------- host orchestration ----------------
extern "C" void kernel_launch(void* const* d_in, const int* in_sizes, int n_in,
                              void* d_out, int out_size, void* d_ws, size_t ws_size,
                              hipStream_t stream)
{
    (void)in_sizes; (void)n_in; (void)out_size; (void)ws_size;

    // d_in layout (setup_inputs insertion order, params flattened recursively):
    // 0 pillars, 1 coors_batch, 2 npoints_per_pillar, 3 pe_w, 4..7 pe_bn{g,b,m,v},
    // backbone blk0 @8+5j (j<4), blk1 @28+5j (j<6), blk2 @58+5j (j<6); layer={w,g,b,m,v}
    // neck @88+5i (i<3) = {w,g,b,m,v}; head: 103 wc,104 bc,105 wr,106 br,107 wd,108 bd
    const float* pil   = (const float*)d_in[0];
    const int*   coors = (const int*)  d_in[1];
    const int*   npp   = (const int*)  d_in[2];

    char* base = (char*)d_ws;
    size_t off = 0;
    auto alloc = [&](size_t bytes) -> char* {
        char* p = base + off;
        off += (bytes + 255) & ~(size_t)255;
        return p;
    };
    float*  pooled = (float*)alloc(sizeof(float) * (size_t)NPIL * CPE);
    int*    map    = (int*)  alloc(sizeof(int)   * (size_t)BSZ * XL * YL);
    // canvas (4,496,432,64) and feat (4,248,216,384) share one region (canvas dies first)
    __bf16* big    = (__bf16*)alloc(sizeof(__bf16) * (size_t)BSZ * HF * WF * 3 * NECK);
    __bf16* p0a    = (__bf16*)alloc(sizeof(__bf16) * (size_t)BSZ * 248 * 216 * 64);
    __bf16* p0b    = (__bf16*)alloc(sizeof(__bf16) * (size_t)BSZ * 248 * 216 * 64);
    __bf16* p1a    = (__bf16*)alloc(sizeof(__bf16) * (size_t)BSZ * 124 * 108 * 128);
    __bf16* p1b    = (__bf16*)alloc(sizeof(__bf16) * (size_t)BSZ * 124 * 108 * 128);
    __bf16* p2a    = (__bf16*)alloc(sizeof(__bf16) * (size_t)BSZ * 62  * 54  * 256);
    __bf16* p2b    = (__bf16*)alloc(sizeof(__bf16) * (size_t)BSZ * 62  * 54  * 256);
    __bf16* pwBase = (__bf16*)alloc(sizeof(__bf16) * 6u * 1024 * 1024);
    __bf16* canvas = big;
    __bf16* feat   = big;

    // ---- pack all weights into B-fragment order (tiny, L2-resident) ----
    size_t pc = 0;
    auto packConv = [&](int wi, int cin, int cs, int cout) -> const __bf16* {
        const int ntiles = cout / 16;
        const int total  = ((cin * 9) / 32) * ntiles * 512;
        __bf16* dst = pwBase + pc; pc += (size_t)total;
        pack_conv_w<<<(total + 255) / 256, 256, 0, stream>>>(
            (const float*)d_in[wi], dst, cin, cs, cout, ntiles, total);
        return dst;
    };
    auto packDec = [&](int wi, int cin, int upup) -> const __bf16* {
        const int total = upup * (cin / 32) * (NECK / 16) * 512;
        __bf16* dst = pwBase + pc; pc += (size_t)total;
        pack_dec_w<<<(total + 255) / 256, 256, 0, stream>>>(
            (const float*)d_in[wi], dst, cin, upup, total);
        return dst;
    };
    auto packFc = [&](int wi, int K, int cout) -> const __bf16* {
        const int ntiles = (cout + 15) / 16;
        const int total  = (K / 32) * ntiles * 512;
        __bf16* dst = pwBase + pc; pc += (size_t)total;
        pack_fc_w<<<(total + 255) / 256, 256, 0, stream>>>(
            (const float*)d_in[wi], dst, K, cout, ntiles, total);
        return dst;
    };

    const __bf16* W[16];
    W[0]  = packConv(8,  64, 6, 64);
    W[1]  = packConv(13, 64, 6, 64);
    W[2]  = packConv(18, 64, 6, 64);
    W[3]  = packConv(23, 64, 6, 64);
    W[4]  = packConv(28, 64, 6, 128);
    W[5]  = packConv(33, 128, 7, 128);
    W[6]  = packConv(38, 128, 7, 128);
    W[7]  = packConv(43, 128, 7, 128);
    W[8]  = packConv(48, 128, 7, 128);
    W[9]  = packConv(53, 128, 7, 128);
    W[10] = packConv(58, 128, 7, 256);
    W[11] = packConv(63, 256, 8, 256);
    W[12] = packConv(68, 256, 8, 256);
    W[13] = packConv(73, 256, 8, 256);
    W[14] = packConv(78, 256, 8, 256);
    W[15] = packConv(83, 256, 8, 256);
    const __bf16* WD0 = packDec(88, 64, 1);
    const __bf16* WD1 = packDec(93, 128, 4);
    const __bf16* WD2 = packDec(98, 256, 16);
    const __bf16* WH0 = packFc(103, 384, 18);
    const __bf16* WH1 = packFc(105, 384, 42);
    const __bf16* WH2 = packFc(107, 384, 12);

    // ---- 1) pillar encoder ----
    pe_encode<<<NPIL, 64, 0, stream>>>(pil, coors, npp,
        (const float*)d_in[3], (const float*)d_in[4], (const float*)d_in[5],
        (const float*)d_in[6], (const float*)d_in[7], pooled);

    // ---- 2) deterministic scatter into zeroed bf16 NHWC canvas ----
    hipMemsetAsync(map, 0xFF, sizeof(int) * (size_t)BSZ * XL * YL, stream);
    hipMemsetAsync(canvas, 0, sizeof(__bf16) * (size_t)BSZ * YL * XL * CPE, stream);
    scatter_argmax<<<(NPIL + 255) / 256, 256, 0, stream>>>(coors, map);
    scatter_write<<<((NPIL * CPE) + 255) / 256, 256, 0, stream>>>(coors, map, pooled, canvas);

    // ---- 3) backbone (2x2 register-blocked waves) ----
#define LAUNCH_CONV(CIN, COUT, STRIDE, xin, widx, bni, yout, HIN, WIN, HOUT, WOUT)          \
    do {                                                                                    \
        const int  mtiles = (BSZ * (HOUT) * (WOUT)) / 16;                                   \
        const long waves  = (long)((mtiles + 1) / 2) * ((COUT) / 32);                       \
        conv3x3_wmma<CIN, COUT, STRIDE><<<(int)((waves + 7) / 8), 256, 0, stream>>>(        \
            xin, W[widx], (const float*)d_in[bni], (const float*)d_in[(bni) + 1],           \
            (const float*)d_in[(bni) + 2], (const float*)d_in[(bni) + 3],                   \
            yout, HIN, WIN, HOUT, WOUT);                                                    \
    } while (0)

    LAUNCH_CONV(64,  64,  2, canvas, 0,  9,  p0a, 496, 432, 248, 216);
    LAUNCH_CONV(64,  64,  1, p0a,    1,  14, p0b, 248, 216, 248, 216);
    LAUNCH_CONV(64,  64,  1, p0b,    2,  19, p0a, 248, 216, 248, 216);
    LAUNCH_CONV(64,  64,  1, p0a,    3,  24, p0b, 248, 216, 248, 216);
    LAUNCH_CONV(64,  128, 2, p0b,    4,  29, p1a, 248, 216, 124, 108);
    LAUNCH_CONV(128, 128, 1, p1a,    5,  34, p1b, 124, 108, 124, 108);
    LAUNCH_CONV(128, 128, 1, p1b,    6,  39, p1a, 124, 108, 124, 108);
    LAUNCH_CONV(128, 128, 1, p1a,    7,  44, p1b, 124, 108, 124, 108);
    LAUNCH_CONV(128, 128, 1, p1b,    8,  49, p1a, 124, 108, 124, 108);
    LAUNCH_CONV(128, 128, 1, p1a,    9,  54, p1b, 124, 108, 124, 108);
    LAUNCH_CONV(128, 256, 2, p1b,    10, 59, p2a, 124, 108, 62,  54);
    LAUNCH_CONV(256, 256, 1, p2a,    11, 64, p2b, 62,  54,  62,  54);
    LAUNCH_CONV(256, 256, 1, p2b,    12, 69, p2a, 62,  54,  62,  54);
    LAUNCH_CONV(256, 256, 1, p2a,    13, 74, p2b, 62,  54,  62,  54);
    LAUNCH_CONV(256, 256, 1, p2b,    14, 79, p2a, 62,  54,  62,  54);
    LAUNCH_CONV(256, 256, 1, p2a,    15, 84, p2b, 62,  54,  62,  54);
#undef LAUNCH_CONV

    // ---- 4) neck deconvs -> feat (4, 248, 216, 384) bf16 NHWC ----
    {
        const long w0 = (long)(((BSZ * 248 * 216 / 16) + 1) / 2) * (NECK / 32);
        deconv_wmma<64, 1><<<dim3((unsigned)((w0 + 7) / 8), 1), 256, 0, stream>>>(
            p0b, WD0, (const float*)d_in[89], (const float*)d_in[90],
            (const float*)d_in[91], (const float*)d_in[92], feat, 248, 216, 0);
        const long w1 = (long)(((BSZ * 124 * 108 / 16) + 1) / 2) * (NECK / 32);
        deconv_wmma<128, 2><<<dim3((unsigned)((w1 + 7) / 8), 4), 256, 0, stream>>>(
            p1b, WD1, (const float*)d_in[94], (const float*)d_in[95],
            (const float*)d_in[96], (const float*)d_in[97], feat, 124, 108, 128);
        const long w2 = (long)(((BSZ * 62 * 54 / 16) + 1) / 2) * (NECK / 32);
        deconv_wmma<256, 4><<<dim3((unsigned)((w2 + 7) / 8), 16), 256, 0, stream>>>(
            p2b, WD2, (const float*)d_in[99], (const float*)d_in[100],
            (const float*)d_in[101], (const float*)d_in[102], feat, 62, 54, 256);
    }

    // ---- 5) heads -> d_out = [cls | reg | dircls], f32 NCHW ----
    float* out = (float*)d_out;
    const long mw = (long)((BSZ * HF * WF) / 16);
    head_wmma<18><<<(int)((mw * 2 + 7) / 8), 256, 0, stream>>>(
        feat, WH0, (const float*)d_in[104], out);
    head_wmma<42><<<(int)((mw * 3 + 7) / 8), 256, 0, stream>>>(
        feat, WH1, (const float*)d_in[106], out + (size_t)BSZ * 18 * HF * WF);
    head_wmma<12><<<(int)((mw * 1 + 7) / 8), 256, 0, stream>>>(
        feat, WH2, (const float*)d_in[108], out + (size_t)BSZ * (18 + 42) * HF * WF);
}
